// Attention_65867618452113
// MI455X (gfx1250) — matrix operand order repaired and checked by reading, not compiled, not run on hardware
//
#include <hip/hip_runtime.h>
#include <hip/hip_bf16.h>

// MI455X fused attention-MLP kernel.
// One workgroup (256 thr = 8 wave32) per batch element b:
//   keys[b] streamed from HBM exactly once (~210MB total -> ~9us floor @23.3TB/s),
//   whole MLP (256->80->40->1) done with v_wmma_f32_16x16x32_bf16 out of LDS,
//   softmax + prob@keys in fp32. bf16 conversion via native fptrunc (RNE).

typedef __attribute__((ext_vector_type(16))) __bf16 v16bf;
typedef __attribute__((ext_vector_type(8)))  float  v8f;

#define S_  200
#define F_  64
#define H1_ 80
#define H2_ 40

__device__ __forceinline__ unsigned short f2bf(float x) {
  union { __bf16 b; unsigned short u; } c; c.b = (__bf16)x; return c.u;
}
__device__ __forceinline__ float bf2f(unsigned short h) {
  union { unsigned u; float f; } v; v.u = ((unsigned)h) << 16;
  return v.f;
}
__device__ __forceinline__ unsigned pk2(float a, float b) {
  return (unsigned)f2bf(a) | ((unsigned)f2bf(b) << 16);
}

union AF { v16bf v; unsigned u[8]; };   // one WMMA A/B fragment (16 bf16 / lane)

__global__ __launch_bounds__(256, 1)
void attn_fused_kernel(const float* __restrict__ query,
                       const float* __restrict__ keys,
                       const int*   __restrict__ seqlen,
                       const float* __restrict__ W1, const float* __restrict__ b1,
                       const float* __restrict__ W2, const float* __restrict__ b2,
                       const float* __restrict__ W3, const float* __restrict__ b3,
                       float* __restrict__ out)
{
  // ---- LDS (~159 KB total) ----
  __shared__ float          keysf[S_ * F_];      // 51200 B, fp32 for final prob@keys
  __shared__ float          qf[F_];              //   256 B
  __shared__ unsigned       w1frag[40 * 256];    // 40960 B: W1 bf16 in B-fragment order
  __shared__ unsigned       w2frag[9 * 256];     //  9216 B: W2 (K pad 96, N pad 48)
  __shared__ unsigned short h1s[208 * 96];       // 39936 B: h1 bf16, rows padded to 208
  __shared__ unsigned short h2s[S_ * 48];        // 19200 B: h2 bf16
  __shared__ float          b1s[H1_];
  __shared__ float          b2s[H2_];
  __shared__ float          w3s[H2_];
  __shared__ float          b3s;
  __shared__ float          scoreL[208];         // scores, then probs
  __shared__ float          red[256];            // reductions / output partials

  const int t    = threadIdx.x;
  const int b    = blockIdx.x;
  const int lane = t & 31;
  const int wv   = t >> 5;

  // ============ Phase 0: stage inputs into LDS ============
  {
    const float4* k4 = (const float4*)(keys + (size_t)b * (S_ * F_));
    float4* d4 = (float4*)keysf;
    for (int i = t; i < (S_ * F_) / 4; i += 256) d4[i] = k4[i];   // coalesced b128

    if (t < F_)  qf[t]  = query[(size_t)b * F_ + t];
    if (t < H1_) b1s[t] = b1[t];
    if (t < H2_) b2s[t] = b2[t];
    if (t < H2_) w3s[t] = W3[t];
    if (t == 0)  b3s    = b3[0];

    // zero h1 K-pad columns 80..95 (8 dwords per row)
    for (int i = t; i < 208 * 8; i += 256) {
      int r = i >> 3, c = i & 7;
      ((unsigned*)h1s)[r * 48 + 40 + c] = 0u;
    }
    // W1 -> B-fragment order: frag(kstep,ntile) lane L vgpr v holds
  //   W1[kb..kb+1][n], n = nt*16+(L&15), kb = ks*32 + (L>>4)*8 + (v>>2)*8 + 2v
    for (int i = t; i < 40 * 256; i += 256) {
      int fi = i >> 8, rem = i & 255;
      int ln = rem >> 3, v = rem & 7;
      int ks = fi / 5, nt = fi % 5;
      int n  = nt * 16 + (ln & 15);
      int kb = ks * 32 + ((ln >> 4) << 3) + ((v >> 2) << 3) + v * 2;
      w1frag[i] = pk2(W1[kb * H1_ + n], W1[(kb + 1) * H1_ + n]);
    }
    // W2 -> B-fragment order, zero padding K>=80 / N>=40
    for (int i = t; i < 9 * 256; i += 256) {
      int fi = i >> 8, rem = i & 255;
      int ln = rem >> 3, v = rem & 7;
      int ks = fi / 3, nt = fi % 3;
      int n  = nt * 16 + (ln & 15);
      int kb = ks * 32 + ((ln >> 4) << 3) + ((v >> 2) << 3) + v * 2;
      float a0 = (kb     < H1_ && n < H2_) ? W2[kb * H2_ + n]       : 0.f;
      float a1 = (kb + 1 < H1_ && n < H2_) ? W2[(kb + 1) * H2_ + n] : 0.f;
      w2frag[i] = pk2(a0, a1);
    }
  }
  __syncthreads();

  // ============ Phase 1: h1 = relu([q,k,q-k,q*k] @ W1 + b1) via WMMA ============
  // Load this lane's q slices once (reused by segs q, q-k, q*k, all M-tiles).
  const int off = (lane >> 4) << 3;               // 0 or 8
  float q0[8], q1[8], q2[8], q3[8];
  {
    const float* qp = &qf[off];
    #pragma unroll
    for (int e = 0; e < 8; ++e) {
      q0[e] = qp[e];      q1[e] = qp[16 + e];
      q2[e] = qp[32 + e]; q3[e] = qp[48 + e];
    }
  }
  // q-only A fragments (K-steps 0,1) are M-tile invariant: build once.
  AF aq0, aq1;
  #pragma unroll
  for (int e = 0; e < 8; ++e) {
    aq0.v[e] = (__bf16)q0[e];  aq0.v[8 + e] = (__bf16)q1[e];
    aq1.v[e] = (__bf16)q2[e];  aq1.v[8 + e] = (__bf16)q3[e];
  }

  for (int mt = wv; mt < 13; mt += 8) {
    const int row = mt * 16 + (lane & 15);
    float k0[8], k1[8], k2[8], k3[8];
    if (row < S_) {
      const float* kp = &keysf[row * F_ + off];
      #pragma unroll
      for (int e = 0; e < 8; ++e) {
        k0[e] = kp[e];      k1[e] = kp[16 + e];
        k2[e] = kp[32 + e]; k3[e] = kp[48 + e];
      }
    } else {
      #pragma unroll
      for (int e = 0; e < 8; ++e) { k0[e] = k1[e] = k2[e] = k3[e] = 0.f; }
    }
    // Remaining A fragments: K-steps 2,3 = keys; 4,5 = q-k; 6,7 = q*k.
    AF a2, a3, a4, a5, a6, a7;
    #pragma unroll
    for (int e = 0; e < 8; ++e) {
      a2.v[e] = (__bf16)k0[e];             a2.v[8 + e] = (__bf16)k1[e];
      a3.v[e] = (__bf16)k2[e];             a3.v[8 + e] = (__bf16)k3[e];
      a4.v[e] = (__bf16)(q0[e] - k0[e]);   a4.v[8 + e] = (__bf16)(q1[e] - k1[e]);
      a5.v[e] = (__bf16)(q2[e] - k2[e]);   a5.v[8 + e] = (__bf16)(q3[e] - k3[e]);
      a6.v[e] = (__bf16)(q0[e] * k0[e]);   a6.v[8 + e] = (__bf16)(q1[e] * k1[e]);
      a7.v[e] = (__bf16)(q2[e] * k2[e]);   a7.v[8 + e] = (__bf16)(q3[e] * k3[e]);
    }
    #pragma unroll
    for (int nt = 0; nt < 5; ++nt) {
      v8f c = {0.f, 0.f, 0.f, 0.f, 0.f, 0.f, 0.f, 0.f};
      #pragma unroll
      for (int ks = 0; ks < 8; ++ks) {
        AF bfrag;
        const uint4* bp = (const uint4*)&w1frag[((ks * 5 + nt) * 32 + lane) * 8];
        ((uint4*)bfrag.u)[0] = bp[0];
        ((uint4*)bfrag.u)[1] = bp[1];
        const v16bf& av = (ks == 0) ? aq0.v : (ks == 1) ? aq1.v :
                          (ks == 2) ? a2.v  : (ks == 3) ? a3.v  :
                          (ks == 4) ? a4.v  : (ks == 5) ? a5.v  :
                          (ks == 6) ? a6.v  : a7.v;
        c = __builtin_amdgcn_wmma_f32_16x16x32_bf16(false, av, false, bfrag.v,
                                                    (short)0, c, false, false);
      }
      const int colg = nt * 16 + (lane & 15);        // < 80
      const float bias = b1s[colg];
      const int rb = mt * 16 + ((lane >> 4) << 3);
      #pragma unroll
      for (int r = 0; r < 8; ++r) {
        float h = c[r] + bias;
        h = h > 0.f ? h : 0.f;
        h1s[(rb + r) * 96 + colg] = f2bf(h);
      }
    }
  }
  __syncthreads();

  // ============ Phase 2: h2 = relu(h1 @ W2 + b2), K=96(pad), N=48(pad) ============
  for (int mt = wv; mt < 13; mt += 8) {
    AF a[3];
    const int row = mt * 16 + (lane & 15);
    const unsigned short* hp = &h1s[row * 96];
    #pragma unroll
    for (int ks = 0; ks < 3; ++ks) {
      const int flo = ks * 32 + ((lane >> 4) << 3);
      uint4 lo = *(const uint4*)&hp[flo];        // 8 contiguous bf16 (K kb..kb+7)
      uint4 hi = *(const uint4*)&hp[flo + 16];   // K kb+16..kb+23
      a[ks].u[0] = lo.x; a[ks].u[1] = lo.y; a[ks].u[2] = lo.z; a[ks].u[3] = lo.w;
      a[ks].u[4] = hi.x; a[ks].u[5] = hi.y; a[ks].u[6] = hi.z; a[ks].u[7] = hi.w;
    }
    #pragma unroll
    for (int nt = 0; nt < 3; ++nt) {
      v8f c = {0.f, 0.f, 0.f, 0.f, 0.f, 0.f, 0.f, 0.f};
      #pragma unroll
      for (int ks = 0; ks < 3; ++ks) {
        AF bfrag;
        const uint4* bp = (const uint4*)&w2frag[((ks * 3 + nt) * 32 + lane) * 8];
        ((uint4*)bfrag.u)[0] = bp[0];
        ((uint4*)bfrag.u)[1] = bp[1];
        c = __builtin_amdgcn_wmma_f32_16x16x32_bf16(false, a[ks].v, false, bfrag.v,
                                                    (short)0, c, false, false);
      }
      const int colg = nt * 16 + (lane & 15);
      if (colg < H2_) {
        const float bias = b2s[colg];
        const int rb = mt * 16 + ((lane >> 4) << 3);
        #pragma unroll
        for (int r = 0; r < 8; ++r) {
          if (rb + r < S_) {
            float h = c[r] + bias;
            h = h > 0.f ? h : 0.f;
            h2s[(rb + r) * 48 + colg] = f2bf(h);
          }
        }
      }
    }
  }
  __syncthreads();

  // ============ Phase 3: score = h2 @ W3 + b3 (one row per thread) ============
  if (t < S_) {
    float acc = b3s;
    const unsigned short* hp = &h2s[t * 48];
    #pragma unroll 8
    for (int j = 0; j < H2_; ++j) acc += bf2f(hp[j]) * w3s[j];
    scoreL[t] = acc;
  }
  __syncthreads();

  // ============ Phase 4: masked softmax over rows ============
  const int L    = seqlen[b];
  const bool uni = (L <= 0);                       // all-masked -> uniform softmax
  const int eff  = uni ? S_ : (L < S_ ? L : S_);
  float s = (t < eff) ? (uni ? 0.f : scoreL[t]) : -3.0e38f;
  red[t] = s;
  __syncthreads();
  #pragma unroll
  for (int off2 = 128; off2 > 0; off2 >>= 1) {
    if (t < off2) red[t] = fmaxf(red[t], red[t + off2]);
    __syncthreads();
  }
  const float m = red[0];
  __syncthreads();
  const float e = (t < eff) ? __expf(s - m) : 0.f;
  red[t] = e;
  __syncthreads();
  #pragma unroll
  for (int off2 = 128; off2 > 0; off2 >>= 1) {
    if (t < off2) red[t] = red[t] + red[t + off2];
    __syncthreads();
  }
  const float inv = 1.f / red[0];
  __syncthreads();
  if (t < S_) scoreL[t] = e * inv;                 // masked rows -> exactly 0
  __syncthreads();

  // ============ Phase 5: out = prob @ keys (fp32) ============
  {
    float acc = 0.f;
    const int q4 = t >> 6, f = t & 63;
    for (int row = q4; row < S_; row += 4) acc += scoreL[row] * keysf[row * F_ + f];
    red[t] = acc;
    __syncthreads();
    if (t < 64)
      out[(size_t)b * F_ + t] = red[t] + red[t + 64] + red[t + 128] + red[t + 192];
  }
}

extern "C" void kernel_launch(void* const* d_in, const int* in_sizes, int n_in,
                              void* d_out, int out_size, void* d_ws, size_t ws_size,
                              hipStream_t stream) {
  const float* query = (const float*)d_in[0];
  const float* keys  = (const float*)d_in[1];
  const int*   slen  = (const int*)d_in[2];
  const float* W1    = (const float*)d_in[3];
  const float* b1    = (const float*)d_in[4];
  const float* W2    = (const float*)d_in[5];
  const float* b2    = (const float*)d_in[6];
  const float* W3    = (const float*)d_in[7];
  const float* b3    = (const float*)d_in[8];
  const int B = in_sizes[0] / F_;                  // query is (B,1,64)
  attn_fused_kernel<<<B, 256, 0, stream>>>(query, keys, slen, W1, b1, W2, b2,
                                           W3, b3, (float*)d_out);
}
